// RNN_40046275067912
// MI455X (gfx1250) — matrix operand order repaired and checked by reading, not compiled, not run on hardware
//
#include <hip/hip_runtime.h>
#include <math.h>

#define BB 2048
#define TT 512
#define BT (BB*TT)

typedef __attribute__((ext_vector_type(16))) __bf16 v16bf;
typedef __attribute__((ext_vector_type(8)))  float  v8f;

union Frag16 {
  v16bf  v;
  uint4  q[2];
  __bf16 e[16];
};

// Native CDNA5 V_TANH_F32 (confirmed in round-2 asm); OCML fallback otherwise.
__device__ __forceinline__ float fast_tanh(float x) {
#if __has_builtin(__builtin_amdgcn_tanhf)
  return __builtin_amdgcn_tanhf(x);
#elif __has_builtin(__builtin_amdgcn_tanh_f32)
  return __builtin_amdgcn_tanh_f32(x);
#else
  return tanhf(x);
#endif
}

// ---------------------------------------------------------------------------
// Layer-1 input projection: K=2, pure elementwise (WMMA pointless at K=2).
// ---------------------------------------------------------------------------
__global__ void xp1_kernel(const float* __restrict__ x, const float* __restrict__ Wih,
                           const float* __restrict__ bih, const float* __restrict__ bhh,
                           float* __restrict__ xp) {
  int idx = blockIdx.x * blockDim.x + threadIdx.x;
  if (idx >= BT * 100) return;
  int bt = idx / 100;
  int n  = idx - bt * 100;
  xp[idx] = x[bt*2] * Wih[n*2] + x[bt*2+1] * Wih[n*2+1] + bih[n] + bhh[n];
}

// ---------------------------------------------------------------------------
// Time-parallel input projection GEMM (layers 2,3):
//   xp[M=BT, N] = Hin[M,K] * Wih[N,K]^T + (b_ih + b_hh)
// bf16 WMMA 16x16x32, fp32 accumulate. Full K blocks unguarded; tail block
// carries per-element K guards; out-of-range N rows clamped (columns
// discarded at the guarded store).
// ---------------------------------------------------------------------------
template<int K, int KPAD, int N, int NT>
__global__ void xp_gemm_kernel(const float* __restrict__ Hin, const float* __restrict__ Wih,
                               const float* __restrict__ bih, const float* __restrict__ bhh,
                               float* __restrict__ xp) {
  const int lane = threadIdx.x & 31;
  const int wave = threadIdx.x >> 5;
  const int sel  = lane >> 4;
  const int l16  = lane & 15;
  const int mbase = blockIdx.x * 128 + wave * 16;
  const int arow  = mbase + l16;       // A: M striped over lanes 0..15 (both halves)

  v8f acc[NT];
#pragma unroll
  for (int nt = 0; nt < NT; ++nt) acc[nt] = 0.0f;

#pragma unroll
  for (int kb = 0; kb < KPAD; kb += 32) {
    Frag16 a;
    const int k0 = kb + sel * 8;       // VGPR0-3: K = k0..k0+7
    const int k1 = kb + 16 + sel * 8;  // VGPR4-7: K = k1..k1+7
    if (kb + 32 <= K) {                // compile-time resolved per unrolled block
#pragma unroll
      for (int j = 0; j < 8; ++j) a.e[j]     = (__bf16)Hin[(size_t)arow * K + k0 + j];
#pragma unroll
      for (int j = 0; j < 8; ++j) a.e[8 + j] = (__bf16)Hin[(size_t)arow * K + k1 + j];
    } else {
#pragma unroll
      for (int j = 0; j < 8; ++j)
        a.e[j]     = (k0 + j < K) ? (__bf16)Hin[(size_t)arow * K + k0 + j] : (__bf16)0.f;
#pragma unroll
      for (int j = 0; j < 8; ++j)
        a.e[8 + j] = (k1 + j < K) ? (__bf16)Hin[(size_t)arow * K + k1 + j] : (__bf16)0.f;
    }

#pragma unroll
    for (int nt = 0; nt < NT; ++nt) {
      Frag16 b;
      const int n  = nt * 16 + l16;
      const int nc = (n < N) ? n : (N - 1);   // clamp: garbage column discarded later
      const int kB = kb + sel * 8;
      if (kb + 32 <= K) {
#pragma unroll
        for (int j = 0; j < 16; ++j) b.e[j] = (__bf16)Wih[nc * K + kB + (j < 8 ? j : j + 8)];
      } else {
#pragma unroll
        for (int j = 0; j < 16; ++j)
          b.e[j] = (kB + (j < 8 ? j : j + 8) < K) ? (__bf16)Wih[nc * K + kB + (j < 8 ? j : j + 8)] : (__bf16)0.f;
      }
      acc[nt] = __builtin_amdgcn_wmma_f32_16x16x32_bf16(false, a.v, false, b.v,
                                                        (short)0, acc[nt], false, false);
    }
  }

#pragma unroll
  for (int nt = 0; nt < NT; ++nt) {
    const int n = nt * 16 + l16;
    if (n < N) {
      const float bias = bih[n] + bhh[n];
#pragma unroll
      for (int r = 0; r < 8; ++r) {
        const int m = mbase + sel * 8 + r;   // D: M = 8*half + vgpr
        xp[(size_t)m * N + n] = acc[nt][r] + bias;
      }
    }
  }
}

// ---------------------------------------------------------------------------
// Sequential recurrence: h_t = tanh(xp_t + h_{t-1} * Whh^T)
// One WG owns MT*16 batch rows for all T steps. Whh fragments hoisted into
// registers once (loop-invariant). h state round-trips through a DOUBLE-
// BUFFERED LDS tile (D-layout -> A-layout), one barrier per step. xp loads
// software-pipelined one step ahead (nontemporal). A-fragment LDS loads
// pipelined one K-block ahead of the WMMAs.
// ---------------------------------------------------------------------------
template<int H, int HPAD, int NT, int MT>
__global__ void scan_kernel(const float* __restrict__ xp, const float* __restrict__ Whh,
                            float* __restrict__ hout) {
  constexpr int ASTR = HPAD + 8;           // padded row stride (bank-conflict break)
  constexpr int KB   = HPAD / 32;          // K blocks per step
  constexpr int HTSZ = MT * 16 * ASTR;
  __shared__ __bf16 ht[2][HTSZ];           // double-buffered h tile

  const int tid  = threadIdx.x;
  const int nthr = NT * 32;
  const int lane = tid & 31;
  const int wave = tid >> 5;               // n-tile index
  const int sel  = lane >> 4;
  const int l16  = lane & 15;
  const int b0   = blockIdx.x * (MT * 16);
  const int n    = wave * 16 + l16;
  const bool nv  = (n < H);
  const int nc   = nv ? n : (H - 1);       // clamped column: unguarded loads

  // Hoist B fragments (Whh row nc) into registers, bf16, zero-padded k >= H.
  Frag16 Bfr[KB];
#pragma unroll
  for (int kb = 0; kb < KB; ++kb) {
#pragma unroll
    for (int j = 0; j < 16; ++j) {
      const int k = kb * 32 + sel * 8 + (j < 8 ? j : j + 8);
      Bfr[kb].e[j] = (k < H) ? (__bf16)Whh[nc * H + k] : (__bf16)0.f;
    }
  }

  // Zero both h-tile buffers: h0 = 0, and pad columns must never be NaN.
  for (int i = tid; i < 2 * HTSZ; i += nthr) ht[0][i] = (__bf16)0.f;
  __syncthreads();

  auto load_xp = [&](int t, float xv[MT][8]) {
#pragma unroll
    for (int mt = 0; mt < MT; ++mt)
#pragma unroll
      for (int r = 0; r < 8; ++r) {
        const int b = b0 + mt * 16 + sel * 8 + r;
        xv[mt][r] = __builtin_nontemporal_load(&xp[((size_t)b * TT + t) * H + nc]);
      }
  };

  float xc[MT][8], xn[MT][8];
  load_xp(0, xc);

  int pr = 0;                              // read buffer index
  for (int t = 0; t < TT; ++t) {
    if (t + 1 < TT) load_xp(t + 1, xn);    // prefetch next step's xp

    v8f acc[MT];
#pragma unroll
    for (int mt = 0; mt < MT; ++mt)
#pragma unroll
      for (int r = 0; r < 8; ++r) acc[mt][r] = xc[mt][r];

    const __bf16* htr = &ht[pr][0];
    Frag16 af[2][MT];
#pragma unroll
    for (int mt = 0; mt < MT; ++mt) {      // prologue: A fragments for kb=0
      const __bf16* p = &htr[(mt * 16 + l16) * ASTR + sel * 8];
      af[0][mt].q[0] = *(const uint4*)(p);
      af[0][mt].q[1] = *(const uint4*)(p + 16);
    }
#pragma unroll
    for (int kb = 0; kb < KB; ++kb) {
      if (kb + 1 < KB) {                   // pipeline next K-block's A loads
#pragma unroll
        for (int mt = 0; mt < MT; ++mt) {
          const __bf16* p = &htr[(mt * 16 + l16) * ASTR + (kb + 1) * 32 + sel * 8];
          af[(kb + 1) & 1][mt].q[0] = *(const uint4*)(p);
          af[(kb + 1) & 1][mt].q[1] = *(const uint4*)(p + 16);
        }
      }
#pragma unroll
      for (int mt = 0; mt < MT; ++mt)
        acc[mt] = __builtin_amdgcn_wmma_f32_16x16x32_bf16(false, af[kb & 1][mt].v,
                                                          false, Bfr[kb].v,
                                                          (short)0, acc[mt], false, false);
    }

#pragma unroll
    for (int mt = 0; mt < MT; ++mt)
#pragma unroll
      for (int r = 0; r < 8; ++r) acc[mt][r] = fast_tanh(acc[mt][r]);

    if (nv) {
#pragma unroll
      for (int mt = 0; mt < MT; ++mt)
#pragma unroll
        for (int r = 0; r < 8; ++r) {
          const int b = b0 + mt * 16 + sel * 8 + r;
          hout[((size_t)b * TT + t) * H + n] = acc[mt][r];
        }
    }

    // Write new h into the other buffer; single barrier per step suffices:
    // the buffer written here was last read before the *previous* barrier.
    __bf16* htw = &ht[pr ^ 1][0];
#pragma unroll
    for (int mt = 0; mt < MT; ++mt)
#pragma unroll
      for (int r = 0; r < 8; ++r)
        htw[(mt * 16 + sel * 8 + r) * ASTR + n] = nv ? (__bf16)acc[mt][r] : (__bf16)0.f;
    __syncthreads();
    pr ^= 1;

#pragma unroll
    for (int mt = 0; mt < MT; ++mt)
#pragma unroll
      for (int r = 0; r < 8; ++r) xc[mt][r] = xn[mt][r];
  }
}

// ---------------------------------------------------------------------------
// Output heads: out = (h3 * Wo1^T + bo1) * Wo2^T + bo2   (30 -> 10 -> 2)
// ---------------------------------------------------------------------------
__global__ void heads_kernel(const float* __restrict__ h3, const float* __restrict__ Wo1,
                             const float* __restrict__ bo1, const float* __restrict__ Wo2,
                             const float* __restrict__ bo2, float* __restrict__ out) {
  const int bt = blockIdx.x * blockDim.x + threadIdx.x;
  if (bt >= BT) return;
  float h[30];
#pragma unroll
  for (int i = 0; i < 30; ++i) h[i] = __builtin_nontemporal_load(&h3[(size_t)bt * 30 + i]);
  float o[10];
#pragma unroll
  for (int j = 0; j < 10; ++j) {
    float s = bo1[j];
#pragma unroll
    for (int i = 0; i < 30; ++i) s += Wo1[j * 30 + i] * h[i];
    o[j] = s;
  }
#pragma unroll
  for (int p = 0; p < 2; ++p) {
    float s = bo2[p];
#pragma unroll
    for (int j = 0; j < 10; ++j) s += Wo2[p * 10 + j] * o[j];
    out[(size_t)bt * 2 + p] = s;
  }
}

// ---------------------------------------------------------------------------
extern "C" void kernel_launch(void* const* d_in, const int* in_sizes, int n_in,
                              void* d_out, int out_size, void* d_ws, size_t ws_size,
                              hipStream_t stream) {
  (void)in_sizes; (void)n_in; (void)out_size; (void)ws_size;
  const float* x    = (const float*)d_in[0];
  const float* Wih1 = (const float*)d_in[1];
  const float* Whh1 = (const float*)d_in[2];
  const float* bih1 = (const float*)d_in[3];
  const float* bhh1 = (const float*)d_in[4];
  const float* Wih2 = (const float*)d_in[5];
  const float* Whh2 = (const float*)d_in[6];
  const float* bih2 = (const float*)d_in[7];
  const float* bhh2 = (const float*)d_in[8];
  const float* Wih3 = (const float*)d_in[9];
  const float* Whh3 = (const float*)d_in[10];
  const float* bih3 = (const float*)d_in[11];
  const float* bhh3 = (const float*)d_in[12];
  const float* Wo1  = (const float*)d_in[13];
  const float* bo1  = (const float*)d_in[14];
  const float* Wo2  = (const float*)d_in[15];
  const float* bo2  = (const float*)d_in[16];
  float* out = (float*)d_out;

  // Workspace: R0 (BT*100 f32), R1 (BT*100 f32), R2 (BT*60 f32) ~ 1.09 GB
  float* R0 = (float*)d_ws;
  float* R1 = R0 + (size_t)BT * 100;
  float* R2 = R1 + (size_t)BT * 100;

  // Layer 1: xp1 -> R0, scan -> h1 in R1
  {
    const int total = BT * 100;
    xp1_kernel<<<(total + 255) / 256, 256, 0, stream>>>(x, Wih1, bih1, bhh1, R0);
  }
  scan_kernel<100, 128, 7, 2><<<BB / 32, 7 * 32, 0, stream>>>(R0, Whh1, R1);

  // Layer 2: xp2 (WMMA GEMM, h1 -> R2), scan -> h2 in R0 (xp1 dead)
  xp_gemm_kernel<100, 128, 60, 4><<<BT / 128, 256, 0, stream>>>(R1, Wih2, bih2, bhh2, R2);
  scan_kernel<60, 64, 4, 2><<<BB / 32, 4 * 32, 0, stream>>>(R2, Whh2, R0);

  // Layer 3: xp3 (WMMA GEMM, h2 -> R2, xp2 dead), scan -> h3 in R1 (h1 dead)
  xp_gemm_kernel<60, 64, 30, 2><<<BT / 128, 256, 0, stream>>>(R0, Wih3, bih3, bhh3, R2);
  scan_kernel<30, 32, 2, 2><<<BB / 32, 2 * 32, 0, stream>>>(R2, Whh3, R1);

  // Heads
  heads_kernel<<<(BT + 255) / 256, 256, 0, stream>>>(R1, Wo1, bo1, Wo2, bo2, out);
}
